// GATv2EmbNet_23673859735756
// MI455X (gfx1250) — matrix operand order repaired and checked by reading, not compiled, not run on hardware
//
#include <hip/hip_runtime.h>

typedef __attribute__((ext_vector_type(16))) _Float16 v16h;
typedef __attribute__((ext_vector_type(8)))  float    v8f;
typedef unsigned int u32x4 __attribute__((ext_vector_type(4)));
typedef int          i32x4 __attribute__((ext_vector_type(4)));
typedef int          i32x8 __attribute__((ext_vector_type(8)));

#define NNODE 25000
#define NEDGE 400000
#define EN_TOT (NEDGE + NNODE)
#define DIMX 64
#define NHEAD 4
#define HHD 256     // H * DH
#define NLAYER 3

// ---------------- WMMA fragment loaders (gfx1250 wave32 layouts) ----------------
__device__ __forceinline__ v16h frag_load_a(const _Float16* __restrict__ rowptr, int lane) {
  const int kb = (lane >> 4) << 3;
  v16h f;
#pragma unroll
  for (int r = 0; r < 4; ++r) {
    f[2*r]     = rowptr[kb + 2*r];
    f[2*r+1]   = rowptr[kb + 2*r + 1];
    f[8+2*r]   = rowptr[16 + kb + 2*r];
    f[8+2*r+1] = rowptr[16 + kb + 2*r + 1];
  }
  return f;
}
__device__ __forceinline__ v16h frag_load_b(const _Float16* __restrict__ colptr, int lane) {
  const int kb = (lane >> 4) << 3;
  v16h f;
#pragma unroll
  for (int r = 0; r < 4; ++r) {
    f[2*r]     = colptr[kb + 2*r];
    f[2*r+1]   = colptr[kb + 2*r + 1];
    f[8+2*r]   = colptr[16 + kb + 2*r];
    f[8+2*r+1] = colptr[16 + kb + 2*r + 1];
  }
  return f;
}

// ---------------- TDM: stage a [rows x K] f16 weight panel into LDS ----------------
__device__ __forceinline__ void tdm_load_2d_f16(const _Float16* gsrc, void* lds_dst,
                                                int rows, int K) {
  unsigned long long ga = (unsigned long long)(size_t)gsrc;
  unsigned int la = (unsigned int)(size_t)lds_dst;   // low 32 bits = LDS byte offset
  u32x4 g0;
  g0[0] = 1u;                                               // count=1, user descriptor
  g0[1] = la;                                               // lds_addr
  g0[2] = (unsigned int)(ga & 0xffffffffu);                 // global_addr[31:0]
  g0[3] = (unsigned int)((ga >> 32) & 0x01ffffffu) | (2u << 30); // addr[56:32] | type=2
  i32x8 g1;
  g1[0] = (1 << 16);                                        // data_size = 1 -> 2 bytes
  g1[1] = (K & 0xffff) << 16;                               // tensor_dim0[15:0]
  g1[2] = ((unsigned)K >> 16) | ((rows & 0xffff) << 16);    // td0[31:16] | td1[15:0]
  g1[3] = ((unsigned)rows >> 16) | ((K & 0xffff) << 16);    // td1[31:16] | tile_dim0
  g1[4] = (rows & 0xffff);                                  // tile_dim1 (tile_dim2=0)
  g1[5] = K;                                                // tensor_dim0_stride[31:0]
  g1[6] = 0;
  g1[7] = 0;
  i32x4 z4 = {0, 0, 0, 0};
#if __clang_major__ >= 23
  i32x8 z8 = {0, 0, 0, 0, 0, 0, 0, 0};
  __builtin_amdgcn_tensor_load_to_lds(g0, g1, z4, z4, z8, 0);
#else
  __builtin_amdgcn_tensor_load_to_lds(g0, g1, z4, z4, 0);
#endif
}

// ---------------- weight convert + transpose: W[K,Nc] f32 -> Wt[Nc,K] f16 ----------------
__global__ void k_transpose_f16(const float* __restrict__ W, _Float16* __restrict__ Wt,
                                int K, int Nc) {
  int i = blockIdx.x * blockDim.x + threadIdx.x;
  if (i >= K * Nc) return;
  int k = i / Nc, n = i - k * Nc;
  Wt[(size_t)n * K + k] = (_Float16)W[i];
}

// ---------------- initial embeddings ----------------
__global__ void k_node_embed(const float* __restrict__ nf, const float* __restrict__ W,
                             const float* __restrict__ b, float* __restrict__ x,
                             _Float16* __restrict__ xh) {
  int i = blockIdx.x * blockDim.x + threadIdx.x;
  if (i >= NNODE * DIMX) return;
  int n = i >> 6, d = i & 63;
  float acc = b[d];
#pragma unroll
  for (int k = 0; k < 16; ++k) acc += nf[n * 16 + k] * W[k * 64 + d];
  acc = fmaxf(acc, 0.f);
  x[i] = acc; xh[i] = (_Float16)acc;
}
__global__ void k_edge_embed(const float* __restrict__ ea, const float* __restrict__ W,
                             const float* __restrict__ b, float* __restrict__ e,
                             _Float16* __restrict__ eh) {
  int i = blockIdx.x * blockDim.x + threadIdx.x;
  if (i >= NEDGE * DIMX) return;
  int n = i >> 6, d = i & 63;
  float acc = b[d];
#pragma unroll
  for (int k = 0; k < 4; ++k) acc += ea[n * 4 + k] * W[k * 64 + d];
  acc = fmaxf(acc, 0.f);
  e[i] = acc; eh[i] = (_Float16)acc;
}

// ---------------- node transforms: xl = x@Wl+bl, xr = x@Wr+br ----------------
__global__ void k_node_gemm(const _Float16* __restrict__ xh, const _Float16* __restrict__ Wlt,
                            const _Float16* __restrict__ Wrt, const float* __restrict__ bl,
                            const float* __restrict__ br, float* __restrict__ xl,
                            float* __restrict__ xr) {
  const int lane = threadIdx.x & 31;
  const int rowBase = blockIdx.x * 16;
  const int cgrp = blockIdx.y * 64;
  const bool full = (rowBase + 16 <= NNODE);      // wave-uniform tile bound
  int arow = rowBase + (lane & 15); if (arow >= NNODE) arow = NNODE - 1;
  const v16h a0 = frag_load_a(xh + (size_t)arow * DIMX, lane);
  const v16h a1 = frag_load_a(xh + (size_t)arow * DIMX + 32, lane);
  const int rhalf = (lane >> 4) * 8;
#pragma unroll
  for (int m = 0; m < 2; ++m) {
    const _Float16* Wt = m ? Wrt : Wlt;
    const float* bb = m ? br : bl;
    float* C = m ? xr : xl;
    for (int ct = 0; ct < 4; ++ct) {
      const int bcol = cgrp + ct * 16 + (lane & 15);
      v16h b0 = frag_load_b(Wt + (size_t)bcol * DIMX, lane);
      v16h b1 = frag_load_b(Wt + (size_t)bcol * DIMX + 32, lane);
      v8f c = {0.f,0.f,0.f,0.f,0.f,0.f,0.f,0.f};
      c = __builtin_amdgcn_wmma_f32_16x16x32_f16(false, a0, false, b0, (short)0, c, false, false);
      c = __builtin_amdgcn_wmma_f32_16x16x32_f16(false, a1, false, b1, (short)0, c, false, false);
      const float bv = bb[bcol];
      float* base = C + (size_t)(rowBase + rhalf) * HHD + bcol;
      if (full) {
#pragma unroll
        for (int r = 0; r < 8; ++r) base[(size_t)r * HHD] = c[r] + bv;
      } else {
#pragma unroll
        for (int r = 0; r < 8; ++r)
          if (rowBase + rhalf + r < NNODE) base[(size_t)r * HHD] = c[r] + bv;
      }
    }
  }
}

// ---------------- self-loop edge attr ----------------
__global__ void k_eloop_scatter(const float* __restrict__ e, const int* __restrict__ col,
                                float* __restrict__ eloop, float* __restrict__ cnt) {
  int i = blockIdx.x * blockDim.x + threadIdx.x;
  if (i >= NEDGE * DIMX) return;
  int eid = i >> 6, d = i & 63;
  int c = col[eid];
  atomicAdd(&eloop[(size_t)c * 64 + d], e[i]);
  if (d == 0) atomicAdd(&cnt[c], 1.f);
}
__global__ void k_eloop_finish(const float* __restrict__ eloop, const float* __restrict__ cnt,
                               _Float16* __restrict__ eh_tail) {
  int i = blockIdx.x * blockDim.x + threadIdx.x;
  if (i >= NNODE * DIMX) return;
  int n = i >> 6;
  eh_tail[i] = (_Float16)(eloop[i] / fmaxf(cnt[n], 1.f));
}

__global__ void k_softmax_init(int* __restrict__ mbuf, float* __restrict__ sbuf, int n) {
  int i = blockIdx.x * blockDim.x + threadIdx.x;
  if (i < n) { mbuf[i] = (int)0x80000000; sbuf[i] = 0.f; }
}

// ---------------- fused ee-GEMM + attention score; weights TDM-staged in LDS ----------------
__global__ void k_score(const _Float16* __restrict__ Ef, const _Float16* __restrict__ Wet,
                        const float* __restrict__ xl, const float* __restrict__ xr,
                        const float* __restrict__ att, const int* __restrict__ row,
                        const int* __restrict__ col, float* __restrict__ score,
                        int* __restrict__ mbuf) {
  __shared__ _Float16 sW[HHD * DIMX];   // 32 KB weight panel, shared by 8 waves
  if (threadIdx.x < 32) {
    tdm_load_2d_f16(Wet, sW, HHD, DIMX);
    __builtin_amdgcn_s_wait_tensorcnt(0);
  }
  __syncthreads();
  const int lane = threadIdx.x & 31;
  const int tb = blockIdx.x * 128 + ((threadIdx.x >> 5) << 4);
  const bool full = (tb + 16 <= EN_TOT);          // wave-uniform
  __builtin_prefetch(Ef + (size_t)(tb + 128) * DIMX, 0, 3);
  int aer = tb + (lane & 15); if (aer >= EN_TOT) aer = EN_TOT - 1;
  const v16h a0 = frag_load_a(Ef + (size_t)aer * DIMX, lane);
  const v16h a1 = frag_load_a(Ef + (size_t)aer * DIMX + 32, lane);
  const int rhalf = (lane >> 4) * 8;
  int rf[8], cf[8];
#pragma unroll
  for (int r = 0; r < 8; ++r) {
    int eid = tb + rhalf + r; if (eid >= EN_TOT) eid = EN_TOT - 1;
    rf[r] = (eid < NEDGE) ? row[eid] : (eid - NEDGE);
    cf[r] = (eid < NEDGE) ? col[eid] : (eid - NEDGE);
  }
  const int lc = lane & 15;
  for (int h = 0; h < NHEAD; ++h) {
    float acc[8];
#pragma unroll
    for (int r = 0; r < 8; ++r) acc[r] = 0.f;
    for (int ct = 0; ct < 4; ++ct) {
      const int colg = h * 64 + ct * 16 + lc;
      v16h b0 = frag_load_b(sW + (size_t)colg * DIMX, lane);
      v16h b1 = frag_load_b(sW + (size_t)colg * DIMX + 32, lane);
      v8f c = {0.f,0.f,0.f,0.f,0.f,0.f,0.f,0.f};
      c = __builtin_amdgcn_wmma_f32_16x16x32_f16(false, a0, false, b0, (short)0, c, false, false);
      c = __builtin_amdgcn_wmma_f32_16x16x32_f16(false, a1, false, b1, (short)0, c, false, false);
      const float av = att[colg];
#pragma unroll
      for (int r = 0; r < 8; ++r) {
        float msg = c[r] + xl[(size_t)rf[r] * HHD + colg] + xr[(size_t)cf[r] * HHD + colg];
        msg = (msg > 0.f) ? msg : 0.2f * msg;
        acc[r] += msg * av;
      }
    }
#pragma unroll
    for (int m = 1; m < 16; m <<= 1)
#pragma unroll
      for (int r = 0; r < 8; ++r) acc[r] += __shfl_xor(acc[r], m, 32);
    if (lc == 0) {
      if (full) {
#pragma unroll
        for (int r = 0; r < 8; ++r) {
          int eid = tb + rhalf + r;
          score[(size_t)eid * NHEAD + h] = acc[r];
          int bi = __float_as_int(acc[r]);
          bi = (bi >= 0) ? bi : (bi ^ 0x7fffffff);
          atomicMax(&mbuf[cf[r] * NHEAD + h], bi);
        }
      } else {
#pragma unroll
        for (int r = 0; r < 8; ++r) {
          int eid = tb + rhalf + r;
          if (eid < EN_TOT) {
            score[(size_t)eid * NHEAD + h] = acc[r];
            int bi = __float_as_int(acc[r]);
            bi = (bi >= 0) ? bi : (bi ^ 0x7fffffff);
            atomicMax(&mbuf[cf[r] * NHEAD + h], bi);
          }
        }
      }
    }
  }
}

__global__ void k_exp(const float* __restrict__ score, const int* __restrict__ mbuf,
                      const int* __restrict__ col, float* __restrict__ aexp,
                      float* __restrict__ sbuf) {
  int i = blockIdx.x * blockDim.x + threadIdx.x;
  if (i >= EN_TOT * NHEAD) return;
  int eid = i >> 2, h = i & 3;
  int c = (eid < NEDGE) ? col[eid] : (eid - NEDGE);
  int mi = mbuf[c * NHEAD + h];
  float mv = __int_as_float((mi >= 0) ? mi : (mi ^ 0x7fffffff));
  float a = __expf(score[i] - mv);
  aexp[i] = a;
  atomicAdd(&sbuf[c * NHEAD + h], a);
}

// one thread = 4 contiguous columns of one edge
__global__ void k_aggregate(const float* __restrict__ aexp, const float* __restrict__ sbuf,
                            const float* __restrict__ xl, const int* __restrict__ row,
                            const int* __restrict__ col, float* __restrict__ outagg) {
  int i = blockIdx.x * blockDim.x + threadIdx.x;
  if (i >= EN_TOT * 64) return;
  int eid = i >> 6; int q = (i & 63) << 2; int h = q >> 6;
  int rfv = (eid < NEDGE) ? row[eid] : (eid - NEDGE);
  int cfv = (eid < NEDGE) ? col[eid] : (eid - NEDGE);
  float w = aexp[(size_t)eid * NHEAD + h] / sbuf[cfv * NHEAD + h];
  const float4 xv = *(const float4*)(xl + (size_t)rfv * HHD + q);
  float* dst = outagg + (size_t)cfv * HHD + q;
  atomicAdd(dst + 0, w * xv.x);
  atomicAdd(dst + 1, w * xv.y);
  atomicAdd(dst + 2, w * xv.z);
  atomicAdd(dst + 3, w * xv.w);
}

// ---------------- fused head-mean + residual + LayerNorm + ReLU ----------------
__global__ void k_node_update(const float* __restrict__ outagg, const float* __restrict__ cb,
                              const float* __restrict__ g, const float* __restrict__ bt,
                              float* __restrict__ x, _Float16* __restrict__ xh) {
  const int n = blockIdx.x;
  const int lane = threadIdx.x;
  float v[2], sum = 0.f;
#pragma unroll
  for (int j = 0; j < 2; ++j) {
    int d = lane * 2 + j;
    const float* o = outagg + (size_t)n * HHD + d;
    float om = 0.25f * (o[0] + o[64] + o[128] + o[192]);
    v[j] = x[(size_t)n * DIMX + d] + om + cb[d];
    sum += v[j];
  }
#pragma unroll
  for (int m = 1; m < 32; m <<= 1) sum += __shfl_xor(sum, m, 32);
  float mu = sum * (1.f / 64.f);
  float var = 0.f;
#pragma unroll
  for (int j = 0; j < 2; ++j) { float d0 = v[j] - mu; var += d0 * d0; }
#pragma unroll
  for (int m = 1; m < 32; m <<= 1) var += __shfl_xor(var, m, 32);
  float rs = rsqrtf(var * (1.f / 64.f) + 1e-5f);
#pragma unroll
  for (int j = 0; j < 2; ++j) {
    int d = lane * 2 + j;
    float y = fmaxf((v[j] - mu) * rs * g[d] + bt[d], 0.f);
    x[(size_t)n * DIMX + d] = y;
    xh[(size_t)n * DIMX + d] = (_Float16)y;
  }
}

// ---------------- edge MLP stage 1 (gather-fused, weights in LDS); E % 128 == 0: no guards ----------------
__global__ void k_eu1(const _Float16* __restrict__ xh, const _Float16* __restrict__ eh,
                      const _Float16* __restrict__ W1t, const float* __restrict__ b1,
                      const int* __restrict__ row, const int* __restrict__ col,
                      _Float16* __restrict__ h1) {
  __shared__ _Float16 sW[DIMX * 192];   // 24 KB
  if (threadIdx.x < 32) {
    tdm_load_2d_f16(W1t, sW, DIMX, 192);
    __builtin_amdgcn_s_wait_tensorcnt(0);
  }
  __syncthreads();
  const int lane = threadIdx.x & 31;
  const int tb = blockIdx.x * 128 + ((threadIdx.x >> 5) << 4);
  const int ae = tb + (lane & 15);
  const int ar = row[ae], ac = col[ae];
  const _Float16* segs[3] = { xh + (size_t)ar * 64, xh + (size_t)ac * 64, eh + (size_t)ae * 64 };
  v16h a[6];
#pragma unroll
  for (int s = 0; s < 6; ++s) a[s] = frag_load_a(segs[s >> 1] + (s & 1) * 32, lane);
  const int rhalf = (lane >> 4) * 8;
  for (int ct = 0; ct < 4; ++ct) {
    const int bcol = ct * 16 + (lane & 15);
    v8f c = {0.f,0.f,0.f,0.f,0.f,0.f,0.f,0.f};
#pragma unroll
    for (int s = 0; s < 6; ++s) {
      v16h b = frag_load_b(sW + (size_t)bcol * 192 + s * 32, lane);
      c = __builtin_amdgcn_wmma_f32_16x16x32_f16(false, a[s], false, b, (short)0, c, false, false);
    }
    const float bv = b1[bcol];
    _Float16* base = h1 + (size_t)(tb + rhalf) * 64 + bcol;
#pragma unroll
    for (int r = 0; r < 8; ++r)
      base[(size_t)r * 64] = (_Float16)fmaxf(c[r] + bv, 0.f);
  }
}

// ---------------- edge MLP stage 2 + residual ReLU; E % 128 == 0: no guards ----------------
__global__ void k_eu2(const _Float16* __restrict__ h1, const _Float16* __restrict__ W2t,
                      const float* __restrict__ b2, float* __restrict__ e,
                      _Float16* __restrict__ eh) {
  __shared__ _Float16 sW[DIMX * DIMX];  // 8 KB
  if (threadIdx.x < 32) {
    tdm_load_2d_f16(W2t, sW, DIMX, DIMX);
    __builtin_amdgcn_s_wait_tensorcnt(0);
  }
  __syncthreads();
  const int lane = threadIdx.x & 31;
  const int tb = blockIdx.x * 128 + ((threadIdx.x >> 5) << 4);
  const int arow = tb + (lane & 15);
  const v16h a0 = frag_load_a(h1 + (size_t)arow * 64, lane);
  const v16h a1 = frag_load_a(h1 + (size_t)arow * 64 + 32, lane);
  const int rhalf = (lane >> 4) * 8;
  for (int ct = 0; ct < 4; ++ct) {
    const int bcol = ct * 16 + (lane & 15);
    v16h b0 = frag_load_b(sW + (size_t)bcol * 64, lane);
    v16h b1 = frag_load_b(sW + (size_t)bcol * 64 + 32, lane);
    v8f c = {0.f,0.f,0.f,0.f,0.f,0.f,0.f,0.f};
    c = __builtin_amdgcn_wmma_f32_16x16x32_f16(false, a0, false, b0, (short)0, c, false, false);
    c = __builtin_amdgcn_wmma_f32_16x16x32_f16(false, a1, false, b1, (short)0, c, false, false);
    const float bv = b2[bcol];
#pragma unroll
    for (int r = 0; r < 8; ++r) {
      size_t idx = (size_t)(tb + rhalf + r) * 64 + bcol;
      float v = fmaxf(e[idx] + c[r] + bv, 0.f);
      e[idx] = v; eh[idx] = (_Float16)v;
    }
  }
}

// =======================================================================================
extern "C" void kernel_launch(void* const* d_in, const int* in_sizes, int n_in,
                              void* d_out, int out_size, void* d_ws, size_t ws_size,
                              hipStream_t stream) {
  const float* nf     = (const float*)d_in[0];
  const int*   ei     = (const int*)d_in[1];
  const float* ea     = (const float*)d_in[2];
  const float* W_node = (const float*)d_in[3];
  const float* b_node = (const float*)d_in[4];
  const float* W_edge = (const float*)d_in[5];
  const float* b_edge = (const float*)d_in[6];
  const float* Wl     = (const float*)d_in[7];
  const float* bl     = (const float*)d_in[8];
  const float* Wr     = (const float*)d_in[9];
  const float* br     = (const float*)d_in[10];
  const float* We     = (const float*)d_in[11];
  const float* att    = (const float*)d_in[12];
  const float* conv_b = (const float*)d_in[13];
  const float* ln_g   = (const float*)d_in[14];
  const float* ln_b   = (const float*)d_in[15];
  const float* euW1   = (const float*)d_in[16];
  const float* eub1   = (const float*)d_in[17];
  const float* euW2   = (const float*)d_in[18];
  const float* eub2   = (const float*)d_in[19];
  const int* row = ei;
  const int* col = ei + NEDGE;

  float* x = (float*)d_out;                         // [N,64]
  float* e = (float*)d_out + (size_t)NNODE * DIMX;  // [E,64]

  char* p = (char*)d_ws;
  auto alloc = [&](size_t bytes) -> void* {
    void* r = (void*)p; p += (bytes + 255) & ~(size_t)255; return r;
  };
  _Float16* xh     = (_Float16*)alloc((size_t)NNODE * DIMX * 2);
  _Float16* ehf    = (_Float16*)alloc((size_t)EN_TOT * DIMX * 2);
  _Float16* Wlt    = (_Float16*)alloc((size_t)NLAYER * HHD * DIMX * 2);
  _Float16* Wrt    = (_Float16*)alloc((size_t)NLAYER * HHD * DIMX * 2);
  _Float16* Wet    = (_Float16*)alloc((size_t)NLAYER * HHD * DIMX * 2);
  _Float16* W1t    = (_Float16*)alloc((size_t)NLAYER * DIMX * 192 * 2);
  _Float16* W2t    = (_Float16*)alloc((size_t)NLAYER * DIMX * DIMX * 2);
  float*    xlb    = (float*)alloc((size_t)NNODE * HHD * 4);
  float*    xrb    = (float*)alloc((size_t)NNODE * HHD * 4);
  float*    score  = (float*)alloc((size_t)EN_TOT * NHEAD * 4);
  float*    aexp   = (float*)alloc((size_t)EN_TOT * NHEAD * 4);
  int*      mbuf   = (int*)alloc((size_t)NNODE * NHEAD * 4);
  float*    sbuf   = (float*)alloc((size_t)NNODE * NHEAD * 4);
  float*    outagg = (float*)alloc((size_t)NNODE * HHD * 4);
  float*    eloop  = (float*)alloc((size_t)NNODE * DIMX * 4);
  float*    cnt    = (float*)alloc((size_t)NNODE * 4);
  _Float16* h1     = (_Float16*)alloc((size_t)NEDGE * DIMX * 2);
  (void)ws_size; (void)in_sizes; (void)n_in; (void)out_size;

  const int TB = 256;
  for (int l = 0; l < NLAYER; ++l) {
    int nwl = DIMX * HHD;
    hipLaunchKernelGGL(k_transpose_f16, dim3((nwl + TB - 1) / TB), dim3(TB), 0, stream,
                       Wl + (size_t)l * nwl, Wlt + (size_t)l * nwl, DIMX, HHD);
    hipLaunchKernelGGL(k_transpose_f16, dim3((nwl + TB - 1) / TB), dim3(TB), 0, stream,
                       Wr + (size_t)l * nwl, Wrt + (size_t)l * nwl, DIMX, HHD);
    hipLaunchKernelGGL(k_transpose_f16, dim3((nwl + TB - 1) / TB), dim3(TB), 0, stream,
                       We + (size_t)l * nwl, Wet + (size_t)l * nwl, DIMX, HHD);
    int nw1 = 192 * DIMX;
    hipLaunchKernelGGL(k_transpose_f16, dim3((nw1 + TB - 1) / TB), dim3(TB), 0, stream,
                       euW1 + (size_t)l * nw1, W1t + (size_t)l * nw1, 192, DIMX);
    int nw2 = DIMX * DIMX;
    hipLaunchKernelGGL(k_transpose_f16, dim3((nw2 + TB - 1) / TB), dim3(TB), 0, stream,
                       euW2 + (size_t)l * nw2, W2t + (size_t)l * nw2, DIMX, DIMX);
  }
  hipLaunchKernelGGL(k_node_embed, dim3((NNODE * DIMX + TB - 1) / TB), dim3(TB), 0, stream,
                     nf, W_node, b_node, x, xh);
  hipLaunchKernelGGL(k_edge_embed, dim3((NEDGE * DIMX + TB - 1) / TB), dim3(TB), 0, stream,
                     ea, W_edge, b_edge, e, ehf);

  const int rowTilesN = (NNODE + 15) / 16;
  const int blkEN     = (EN_TOT + 127) / 128;
  const int blkE      = NEDGE / 128;              // exact

  for (int l = 0; l < NLAYER; ++l) {
    const _Float16* Wlt_l = Wlt + (size_t)l * HHD * DIMX;
    const _Float16* Wrt_l = Wrt + (size_t)l * HHD * DIMX;
    const _Float16* Wet_l = Wet + (size_t)l * HHD * DIMX;
    const _Float16* W1t_l = W1t + (size_t)l * DIMX * 192;
    const _Float16* W2t_l = W2t + (size_t)l * DIMX * DIMX;

    hipLaunchKernelGGL(k_node_gemm, dim3(rowTilesN, 4), dim3(32), 0, stream,
                       xh, Wlt_l, Wrt_l, bl + (size_t)l * HHD, br + (size_t)l * HHD, xlb, xrb);

    hipMemsetAsync(eloop, 0, (size_t)NNODE * DIMX * 4, stream);
    hipMemsetAsync(cnt, 0, (size_t)NNODE * 4, stream);
    hipLaunchKernelGGL(k_eloop_scatter, dim3((NEDGE * DIMX + TB - 1) / TB), dim3(TB), 0, stream,
                       e, col, eloop, cnt);
    hipLaunchKernelGGL(k_eloop_finish, dim3((NNODE * DIMX + TB - 1) / TB), dim3(TB), 0, stream,
                       eloop, cnt, ehf + (size_t)NEDGE * DIMX);

    hipLaunchKernelGGL(k_softmax_init, dim3((NNODE * NHEAD + TB - 1) / TB), dim3(TB), 0, stream,
                       mbuf, sbuf, NNODE * NHEAD);
    hipMemsetAsync(outagg, 0, (size_t)NNODE * HHD * 4, stream);

    hipLaunchKernelGGL(k_score, dim3(blkEN), dim3(256), 0, stream,
                       ehf, Wet_l, xlb, xrb, att + (size_t)l * NHEAD * DIMX,
                       row, col, score, mbuf);
    hipLaunchKernelGGL(k_exp, dim3((EN_TOT * NHEAD + TB - 1) / TB), dim3(TB), 0, stream,
                       score, mbuf, col, aexp, sbuf);
    hipLaunchKernelGGL(k_aggregate, dim3((EN_TOT * 64 + TB - 1) / TB), dim3(TB), 0, stream,
                       aexp, sbuf, xlb, row, col, outagg);
    hipLaunchKernelGGL(k_node_update, dim3(NNODE), dim3(32), 0, stream,
                       outagg, conv_b + (size_t)l * DIMX, ln_g + (size_t)l * DIMX,
                       ln_b + (size_t)l * DIMX, x, xh);
    hipLaunchKernelGGL(k_eu1, dim3(blkE), dim3(256), 0, stream,
                       xh, ehf, W1t_l, eub1 + (size_t)l * DIMX, row, col, h1);
    hipLaunchKernelGGL(k_eu2, dim3(blkE), dim3(256), 0, stream,
                       h1, W2t_l, eub2 + (size_t)l * DIMX, e, ehf);
  }
}
